// GatedAttention_30288109371488
// MI455X (gfx1250) — matrix operand order repaired and checked by reading, not compiled, not run on hardware
//
#include <hip/hip_runtime.h>
#include <hip/hip_bf16.h>
#include <math.h>

// ---------------------------------------------------------------------------
// GatedAttention for MI455X (gfx1250, wave32, WMMA).
// B=4, N=2048, D=256, H=4, DH=64.  All GEMMs via v_wmma_f32_16x16x32_f16.
// ---------------------------------------------------------------------------

typedef _Float16 v8h  __attribute__((ext_vector_type(8)));
typedef _Float16 v16h __attribute__((ext_vector_type(16)));
typedef float    v8f  __attribute__((ext_vector_type(8)));

#define NB   4
#define NN   2048
#define ND   256
#define NH   4
#define NDH  64
#define ROWS (NB * NN)   // 8192

static __device__ inline v8f wmma32(v16h a, v16h b, v8f c) {
  // D = A(16x32) * B(32x16) + C, f16 in / f32 out.
  return __builtin_amdgcn_wmma_f32_16x16x32_f16(
      /*neg_a=*/false, a, /*neg_b=*/false, b,
      /*c_mod=*/(short)0, c, /*reuse_a=*/false, /*reuse_b=*/false);
}

// A operand (16x32 f16, row-major source, leading dim ld in halfs).
// Lane layout: lane m (lo): K = kc..kc+15;
//              lane m+16  : K = kc+16..kc+31.
static __device__ inline v16h load_a(const _Float16* p, int row, int ld, int kc, int hi) {
  const _Float16* base = p + row * ld + kc + hi * 16;
  v8h lo = *(const v8h*)(base);
  v8h hg = *(const v8h*)(base + 8);
  return __builtin_shufflevector(lo, hg, 0,1,2,3,4,5,6,7,8,9,10,11,12,13,14,15);
}

// B operand (32x16 f16) from a column-major [cols][K] buffer.
// Lane n holds column n, K rows kc..kc+15 (lo half) / kc+16..kc+31 (hi half).
static __device__ inline v16h load_b(const _Float16* p, int col, int ld, int kc, int hi) {
  return *(const v16h*)(p + col * ld + kc + hi * 16);
}

// ---------------------------------------------------------------------------
// 0) Convert + transpose weights to f16 column-major ([out_col][k]).
// ---------------------------------------------------------------------------
__global__ void prep_weights(const float* __restrict__ w_qkv,
                             const float* __restrict__ w_out,
                             const float* __restrict__ w_gate,
                             _Float16* __restrict__ wqkvT,
                             _Float16* __restrict__ woutT,
                             _Float16* __restrict__ wgateT) {
  int idx = blockIdx.x * blockDim.x + threadIdx.x;
  if (idx < 3 * ND * ND * 3) { /* unreachable guard pattern, real ranges below */ }
  if (idx < 768 * 256) {                    // wqkvT[j*256+k] = w_qkv[k*768+j]
    int j = idx >> 8, k = idx & 255;
    wqkvT[idx] = (_Float16)w_qkv[k * 768 + j];
  } else if (idx < 768 * 256 + 256 * 256) { // woutT
    int t = idx - 768 * 256;
    int j = t >> 8, k = t & 255;
    woutT[t] = (_Float16)w_out[k * 256 + j];
  } else if (idx < 768 * 256 + 2 * 256 * 256) { // wgateT
    int t = idx - (768 * 256 + 256 * 256);
    int j = t >> 8, k = t & 255;
    wgateT[t] = (_Float16)w_gate[k * 256 + j];
  }
}

// ---------------------------------------------------------------------------
// 1) Zero-centered RMSNorm: h = (x - mean) / rms * g, stored f16. One row/block.
// ---------------------------------------------------------------------------
__global__ __launch_bounds__(256) void norm_kernel(const float* __restrict__ x,
                                                   const float* __restrict__ g,
                                                   _Float16* __restrict__ h16) {
  __shared__ float red[8];
  int row = blockIdx.x, tid = threadIdx.x;
  float v = x[row * ND + tid];
  float s = v;
  #pragma unroll
  for (int o = 16; o > 0; o >>= 1) s += __shfl_xor(s, o, 32);
  if ((tid & 31) == 0) red[tid >> 5] = s;
  __syncthreads();
  float tot = 0.f;
  #pragma unroll
  for (int i = 0; i < 8; ++i) tot += red[i];
  float mean = tot * (1.0f / ND);
  float c = v - mean;
  float sq = c * c;
  #pragma unroll
  for (int o = 16; o > 0; o >>= 1) sq += __shfl_xor(sq, o, 32);
  __syncthreads();
  if ((tid & 31) == 0) red[tid >> 5] = sq;
  __syncthreads();
  float tot2 = 0.f;
  #pragma unroll
  for (int i = 0; i < 8; ++i) tot2 += red[i];
  float rms = sqrtf(tot2 * (1.0f / ND) + 1e-8f);
  h16[row * ND + tid] = (_Float16)(c / rms * g[tid]);
}

// ---------------------------------------------------------------------------
// 2) qkv = h @ w_qkv + b, RoPE on q,k; scatter q,k row-major / v transposed.
//    One wave per (16-row tile, 64-col strip).  strip s -> m3 = s/4, head = s%4.
// ---------------------------------------------------------------------------
__global__ __launch_bounds__(32) void qkv_kernel(const _Float16* __restrict__ h16,
                                                 const _Float16* __restrict__ wqkvT,
                                                 const float* __restrict__ b_qkv,
                                                 const float* __restrict__ freqs,
                                                 _Float16* __restrict__ q16,
                                                 _Float16* __restrict__ k16,
                                                 _Float16* __restrict__ vt16) {
  int lid = threadIdx.x, n = lid & 15, hi = lid >> 4;
  int rt = blockIdx.x;        // 0..511 row tiles
  int s  = blockIdx.y;        // 0..11 column strips of 64
  int m3 = s >> 2, head = s & 3;

  v8f acc[4];
  #pragma unroll
  for (int t = 0; t < 4; ++t) acc[t] = (v8f){0,0,0,0,0,0,0,0};

  for (int kc = 0; kc < ND; kc += 32) {
    v16h a = load_a(h16, rt * 16 + n, ND, kc, hi);
    #pragma unroll
    for (int t = 0; t < 4; ++t) {
      v16h b = load_b(wqkvT, s * 64 + t * 16 + n, ND, kc, hi);
      acc[t] = wmma32(a, b, acc[t]);
    }
  }

  #pragma unroll
  for (int t = 0; t < 4; ++t) {
    int dh  = t * 16 + n;           // 0..63 within head
    int col = s * 64 + dh;          // global qkv column = m3*256 + head*64 + dh
    float bias = b_qkv[col];
    #pragma unroll
    for (int r = 0; r < 8; ++r) {
      float val = acc[t][r] + bias;
      int row = rt * 16 + r + hi * 8;
      int bb = row >> 11, pos = row & (NN - 1);
      if (m3 < 2) {
        // RoPE: pairs (2i, 2i+1) live in adjacent lanes.
        float partner = __shfl_xor(val, 1, 32);
        const float* f = freqs + ((pos << 5) + (dh >> 1)) * 2;
        float cs = f[0], sn = f[1];
        val = (dh & 1) ? (val * cs + partner * sn) : (val * cs - partner * sn);
        _Float16* dst = (m3 == 0) ? q16 : k16;
        dst[(((bb << 2) + head) * NN + pos) * NDH + dh] = (_Float16)val;
      } else {
        vt16[(((bb << 2) + head) * NDH + dh) * NN + pos] = (_Float16)val;
      }
    }
  }
}

// ---------------------------------------------------------------------------
// 3) Flash attention (transposed form, register-resident).
//    One wave per (b,h, 16-query tile).  Per 32-key chunk: 4 WMMA for
//    S^T = K*(Q^T/8), online softmax, 4 WMMA for Y^T += Vt*P^T.
// ---------------------------------------------------------------------------
__global__ __launch_bounds__(32) void attn_kernel(const _Float16* __restrict__ q16,
                                                  const _Float16* __restrict__ k16,
                                                  const _Float16* __restrict__ vt16,
                                                  _Float16* __restrict__ y16) {
  int lid = threadIdx.x, n = lid & 15, hi = lid >> 4;
  int qt = blockIdx.x & 127;
  int bh = blockIdx.x >> 7;
  const _Float16* Q  = q16  + (size_t)bh * NN * NDH;
  const _Float16* K  = k16  + (size_t)bh * NN * NDH;
  const _Float16* Vt = vt16 + (size_t)bh * NDH * NN;

  int q = qt * 16 + n;
  // Q^T as B operand, loop-invariant; fold the 1/sqrt(DH)=0.125 scale in.
  v16h bq0 = *(const v16h*)(Q + q * NDH + hi * 16);
  v16h bq1 = *(const v16h*)(Q + q * NDH + 32 + hi * 16);
  #pragma unroll
  for (int i = 0; i < 16; ++i) { bq0[i] *= (_Float16)0.125f; bq1[i] *= (_Float16)0.125f; }

  v8f yacc[4];
  #pragma unroll
  for (int t = 0; t < 4; ++t) yacc[t] = (v8f){0,0,0,0,0,0,0,0};
  float m_i = -1e30f, l_i = 0.f;
  const v8f z = (v8f){0,0,0,0,0,0,0,0};

  for (int kv = 0; kv < NN; kv += 32) {
    // S^T tiles: keys kv..kv+15 (st0) and kv+16..kv+31 (st1), 16 queries.
    v16h a00 = load_a(K, kv + n,      NDH, 0,  hi);
    v16h a01 = load_a(K, kv + n,      NDH, 32, hi);
    v16h a10 = load_a(K, kv + 16 + n, NDH, 0,  hi);
    v16h a11 = load_a(K, kv + 16 + n, NDH, 32, hi);
    v8f st0 = wmma32(a00, bq0, z); st0 = wmma32(a01, bq1, st0);
    v8f st1 = wmma32(a10, bq0, z); st1 = wmma32(a11, bq1, st1);

    // Per-query (per-lane) online softmax; combine halves with shfl_xor(16).
    float mx = -1e30f;
    #pragma unroll
    for (int r = 0; r < 8; ++r) mx = fmaxf(mx, fmaxf(st0[r], st1[r]));
    mx = fmaxf(mx, __shfl_xor(mx, 16, 32));
    float m_new = fmaxf(m_i, mx);
    float alpha = __expf(m_i - m_new);

    float e0[8], e1[8], sum = 0.f;
    #pragma unroll
    for (int r = 0; r < 8; ++r) {
      e0[r] = __expf(st0[r] - m_new);
      e1[r] = __expf(st1[r] - m_new);
      sum += e0[r] + e1[r];
    }
    sum += __shfl_xor(sum, 16, 32);
    l_i = l_i * alpha + sum;
    m_i = m_new;
    #pragma unroll
    for (int t = 0; t < 4; ++t)
      #pragma unroll
      for (int r = 0; r < 8; ++r) yacc[t][r] *= alpha;

    // Assemble P^T as B operand: lane lo needs keys 0..15, lane hi 16..31.
    v16h pb;
    #pragma unroll
    for (int r = 0; r < 8; ++r) {
      float x0 = __shfl_xor(e0[r], 16, 32);  // lo lane <- keys 8..15
      float x1 = __shfl_xor(e1[r], 16, 32);  // hi lane <- keys 16..23
      pb[r]     = (_Float16)(hi ? x1 : e0[r]);
      pb[8 + r] = (_Float16)(hi ? e1[r] : x0);
    }

    // Y^T += Vt(16dh x 32keys) * P^T(32keys x 16q), 4 dh tiles.
    #pragma unroll
    for (int t = 0; t < 4; ++t) {
      v16h av = load_a(Vt, t * 16 + n, NN, kv, hi);
      yacc[t] = wmma32(av, pb, yacc[t]);
    }
  }

  float inv = 1.0f / l_i;
  int bb = bh >> 2, head = bh & 3;
  #pragma unroll
  for (int t = 0; t < 4; ++t)
    #pragma unroll
    for (int r = 0; r < 8; ++r) {
      int dh = t * 16 + r + hi * 8;
      y16[((size_t)bb * NN + q) * ND + head * NDH + dh] = (_Float16)(yacc[t][r] * inv);
    }
}

// ---------------------------------------------------------------------------
// 4) out = x + sigmoid(h@w_gate + b_gate) * (y@w_out + b_out)
//    One wave per 16x16 output tile; two WMMA chains (out-proj + gate).
// ---------------------------------------------------------------------------
__global__ __launch_bounds__(32) void final_kernel(const float* __restrict__ x,
                                                   const _Float16* __restrict__ h16,
                                                   const _Float16* __restrict__ y16,
                                                   const _Float16* __restrict__ woutT,
                                                   const _Float16* __restrict__ wgateT,
                                                   const float* __restrict__ b_out,
                                                   const float* __restrict__ b_gate,
                                                   float* __restrict__ out) {
  int lid = threadIdx.x, n = lid & 15, hi = lid >> 4;
  int rt = blockIdx.x;   // 0..511
  int ct = blockIdx.y;   // 0..15
  v8f ao = (v8f){0,0,0,0,0,0,0,0};
  v8f ag = (v8f){0,0,0,0,0,0,0,0};
  for (int kc = 0; kc < ND; kc += 32) {
    v16h ay = load_a(y16, rt * 16 + n, ND, kc, hi);
    v16h ah = load_a(h16, rt * 16 + n, ND, kc, hi);
    v16h bo = load_b(woutT,  ct * 16 + n, ND, kc, hi);
    v16h bg = load_b(wgateT, ct * 16 + n, ND, kc, hi);
    ao = wmma32(ay, bo, ao);
    ag = wmma32(ah, bg, ag);
  }
  int j = ct * 16 + n;
  float bo_ = b_out[j], bg_ = b_gate[j];
  #pragma unroll
  for (int r = 0; r < 8; ++r) {
    int row = rt * 16 + r + hi * 8;
    float zo = ao[r] + bo_;
    float zg = ag[r] + bg_;
    float gate = 1.0f / (1.0f + __expf(-zg));
    out[row * ND + j] = x[row * ND + j] + gate * zo;
  }
}

// ---------------------------------------------------------------------------
extern "C" void kernel_launch(void* const* d_in, const int* in_sizes, int n_in,
                              void* d_out, int out_size, void* d_ws, size_t ws_size,
                              hipStream_t stream) {
  const float* x      = (const float*)d_in[0];
  const float* freqs  = (const float*)d_in[1];
  const float* g      = (const float*)d_in[2];
  const float* w_qkv  = (const float*)d_in[3];
  const float* b_qkv  = (const float*)d_in[4];
  const float* w_out  = (const float*)d_in[5];
  const float* b_out  = (const float*)d_in[6];
  const float* w_gate = (const float*)d_in[7];
  const float* b_gate = (const float*)d_in[8];
  float* out = (float*)d_out;

  const size_t TOK = (size_t)ROWS * ND;          // 2,097,152 elements
  _Float16* h16    = (_Float16*)d_ws;
  _Float16* q16    = h16  + TOK;
  _Float16* k16    = q16  + TOK;
  _Float16* vt16   = k16  + TOK;
  _Float16* y16    = vt16 + TOK;
  _Float16* wqkvT  = y16  + TOK;                 // 196608
  _Float16* woutT  = wqkvT + 768 * 256;          // 65536
  _Float16* wgateT = woutT + 256 * 256;          // 65536  (total ~20.6 MB)

  prep_weights<<<dim3(1280), dim3(256), 0, stream>>>(w_qkv, w_out, w_gate,
                                                     wqkvT, woutT, wgateT);
  norm_kernel<<<dim3(ROWS), dim3(256), 0, stream>>>(x, g, h16);
  qkv_kernel<<<dim3(ROWS / 16, 12), dim3(32), 0, stream>>>(h16, wqkvT, b_qkv, freqs,
                                                           q16, k16, vt16);
  attn_kernel<<<dim3(NB * NH * (NN / 16)), dim3(32), 0, stream>>>(q16, k16, vt16, y16);
  final_kernel<<<dim3(ROWS / 16, 16), dim3(32), 0, stream>>>(x, h16, y16, woutT, wgateT,
                                                             b_out, b_gate, out);
}
